// WindowAttention_2894807957857
// MI455X (gfx1250) — matrix operand order repaired and checked by reading, not compiled, not run on hardware
//
#include <hip/hip_runtime.h>
#include <hip/hip_bf16.h>

typedef __attribute__((ext_vector_type(16))) _Float16 v16h;
typedef __attribute__((ext_vector_type(8)))  _Float16 v8h;
typedef __attribute__((ext_vector_type(8)))  float    v8f;
typedef __attribute__((ext_vector_type(4)))  unsigned int su4;
typedef __attribute__((ext_vector_type(8)))  unsigned int su8;
typedef unsigned long long ull;

#define B_WIN   2048
#define N_TOK   49
#define C_DIM   384
#define NHEADS  12
#define HD      32
#define M_TOTAL (B_WIN * N_TOK)        // 100352 (divisible by 16)
#define MT      (M_TOTAL / 16)         // 6272
#define NT64    (C_DIM / 64)           // 6 column tiles of 64
#define TILES64 (MT * NT64)            // 37632 wave-tiles
#define WAVES_PER_BLOCK 8
#define GEMM_BLOCKS (TILES64 / WAVES_PER_BLOCK)   // 4704
#define WELEMS  (C_DIM * C_DIM)        // 147456
#define SCALE_Q 0.17677669529663687f   // 32^-0.5

__device__ __forceinline__ v8f wmma_f16(v16h a, v16h b, v8f c) {
  // D = A(16x32 f16) x B(32x16 f16) + C(16x16 f32)
  return __builtin_amdgcn_wmma_f32_16x16x32_f16(
      false, a, false, b, (short)0, c, false, false);
}

// ---------------------------------------------------------------------------
// Kernel 0: one-time weight conversion f32 -> f16 (weights stay L2-resident;
// removes all v_cvt on the B-operand stream of every GEMM below).
// ---------------------------------------------------------------------------
__global__ __launch_bounds__(256) void weights_to_f16(
    const float* __restrict__ Wq, const float* __restrict__ Wk,
    const float* __restrict__ Wv, const float* __restrict__ Wp,
    _Float16* __restrict__ Wh)
{
  const int z = blockIdx.z;
  const float* src = (z == 0) ? Wq : (z == 1) ? Wk : (z == 2) ? Wv : Wp;
  const int i = blockIdx.x * 256 + threadIdx.x;
  Wh[(size_t)z * WELEMS + i] = (_Float16)src[i];
}

// ---------------------------------------------------------------------------
// Kernel 1: fused QKV projections; blockIdx.z selects q/k/v.
// One wave computes a 16x64 tile: per k-step 1 A-frag feeds 4 WMMAs.
// Output: row-major [m][C] f16 (coalesced stores); q pre-scaled by hd^-0.5.
// ---------------------------------------------------------------------------
__global__ __launch_bounds__(256) void qkv_proj_wmma(
    const float* __restrict__ q, const float* __restrict__ k, const float* __restrict__ v,
    const _Float16* __restrict__ WhAll,
    const float* __restrict__ bq, const float* __restrict__ bk, const float* __restrict__ bv,
    _Float16* __restrict__ qh, _Float16* __restrict__ kh, _Float16* __restrict__ vh)
{
  const int which = blockIdx.z;
  const float*    X   = (which == 0) ? q  : (which == 1) ? k  : v;
  const _Float16* Wh  = WhAll + (size_t)which * WELEMS;
  const float*    bia = (which == 0) ? bq : (which == 1) ? bk : bv;
  _Float16*       dst = (which == 0) ? qh : (which == 1) ? kh : vh;
  const float oscale  = (which == 0) ? SCALE_Q : 1.0f;

  const int lane = threadIdx.x & 31;
  const int wave = threadIdx.x >> 5;
  const int tile = blockIdx.x * WAVES_PER_BLOCK + wave;
  const int mt = tile / NT64, nt = tile % NT64;
  const int m0 = mt * 16, n0 = nt * 64;
  const int hi  = lane >> 4;
  const int l15 = lane & 15;

  const float* arow = X + (size_t)(m0 + l15) * C_DIM + 16 * hi;
  const _Float16* brow0 = Wh + (size_t)(n0 +      l15) * C_DIM + 16 * hi;
  const _Float16* brow1 = Wh + (size_t)(n0 + 16 + l15) * C_DIM + 16 * hi;
  const _Float16* brow2 = Wh + (size_t)(n0 + 32 + l15) * C_DIM + 16 * hi;
  const _Float16* brow3 = Wh + (size_t)(n0 + 48 + l15) * C_DIM + 16 * hi;

  v8f acc0 = {}, acc1 = {}, acc2 = {}, acc3 = {};
  for (int k0 = 0; k0 < C_DIM; k0 += 32) {
    __builtin_prefetch(arow + k0 + 32, 0, 3);     // WGP-scope prefetch
    const v8f af0 = *(const v8f*)(arow + k0);     // 32B-aligned b128 pair
    const v8f af1 = *(const v8f*)(arow + k0 + 8);
    v16h a;
#pragma unroll
    for (int j = 0; j < 8; ++j) { a[j] = (_Float16)af0[j]; a[8 + j] = (_Float16)af1[j]; }
    acc0 = wmma_f16(a, *(const v16h*)(brow0 + k0), acc0);
    acc1 = wmma_f16(a, *(const v16h*)(brow1 + k0), acc1);
    acc2 = wmma_f16(a, *(const v16h*)(brow2 + k0), acc2);
    acc3 = wmma_f16(a, *(const v16h*)(brow3 + k0), acc3);
  }

#pragma unroll
  for (int t = 0; t < 4; ++t) {
    const v8f accv = (t == 0) ? acc0 : (t == 1) ? acc1 : (t == 2) ? acc2 : acc3;
    const int col = n0 + 16 * t + l15;
    const float bval = bia[col];
#pragma unroll
    for (int r = 0; r < 8; ++r) {
      const int m = m0 + r + 8 * hi;
      dst[(size_t)m * C_DIM + col] = (_Float16)((accv[r] + bval) * oscale);
    }
  }
}

// ---------------------------------------------------------------------------
// Kernel 2: windowed attention. One block (4 waves / 128 threads) per (b,h).
//   Q  -> LDS via global_load_async_to_lds_b128 (ASYNCcnt path)
//   K  -> LDS via Tensor Data Mover descriptor  (TENSORcnt path)
//   V  -> LDS transposed through VGPRs
// S = Q K^T via WMMA, softmax + rel-bias, O = P V via WMMA.
// ---------------------------------------------------------------------------
__global__ __launch_bounds__(128) void win_attn_wmma(
    const _Float16* __restrict__ qh, const _Float16* __restrict__ kh,
    const _Float16* __restrict__ vh, const float* __restrict__ bias_table,
    _Float16* __restrict__ Xo)
{
  __shared__ alignas(64) _Float16 Qs[64 * HD];    // 4 KB, row-major 64x32
  __shared__ alignas(64) _Float16 Ks[64 * HD];    // 4 KB
  __shared__ alignas(64) _Float16 Vt[HD * 64];    // 4 KB, transposed [d][tok]
  __shared__ alignas(64) float    Ss[64 * 64];    // 16 KB logits / exp scratch
  __shared__ alignas(64) _Float16 Ps[64 * 64];    // 8 KB probs (f16)

  const int bh   = blockIdx.x;            // b*12 + h
  const int head = bh % NHEADS;
  const int bwin = bh / NHEADS;
  // row-major [m][C]; head h occupies 64 contiguous bytes per token row
  const size_t gbase = ((size_t)bwin * N_TOK) * C_DIM + (size_t)head * HD;

  // Zero-pad rows 49..63 (DS stores; disjoint from DMA/async-written region)
  for (int i = threadIdx.x; i < (64 - N_TOK) * HD; i += 128) {
    Qs[N_TOK * HD + i] = (_Float16)0.f;
    Ks[N_TOK * HD + i] = (_Float16)0.f;
  }

  // ---- Q: async global->LDS, 49 rows x 64B = 196 16-byte chunks ----
  {
    const unsigned lQ = (unsigned)(uintptr_t)(&Qs[0]);
    const ull gQ = (ull)(uintptr_t)(qh + gbase);
    for (int c = threadIdx.x; c < 4 * N_TOK; c += 128) {
      const int r = c >> 2, s = c & 3;
      const unsigned goff = (unsigned)(r * (C_DIM * 2) + s * 16);  // bytes
      const unsigned la = lQ + (unsigned)(c * 16);
      const ull      ga = gQ + goff;
      asm volatile("global_load_async_to_lds_b128 %0, %1, off"
                   :: "v"(la), "v"(ga) : "memory");
    }
  }

  // ---- K: Tensor Data Mover. 2D tile 32x49 f16, row stride 384 elems. ----
  {
    const unsigned lK = (unsigned)(uintptr_t)(&Ks[0]);
    const ull gK = (ull)(uintptr_t)(kh + gbase);
    su4 g0;
    g0[0] = 1u;                                           // count=1, user D#
    g0[1] = lK;                                           // lds_addr (bytes)
    g0[2] = (unsigned)(gK & 0xffffffffu);                 // global_addr[31:0]
    g0[3] = (unsigned)((gK >> 32) & 0x01ffffffu)          // global_addr[56:32]
            | (2u << 30);                                 // type = 2 (image)
    su8 g1;
    g1[0] = 1u << 16;                                     // data_size = 2 bytes
    g1[1] = (unsigned)HD << 16;                           // tensor_dim0 = 32
    g1[2] = (unsigned)N_TOK << 16;                        // tensor_dim1 = 49
    g1[3] = (unsigned)HD << 16;                           // tile_dim0 = 32
    g1[4] = (unsigned)N_TOK;                              // tile_dim1 = 49
    g1[5] = (unsigned)C_DIM;                              // dim0_stride = 384
    g1[6] = 0u;
    g1[7] = 0u;
    if ((threadIdx.x >> 5) == 0) {                        // one wave issues DMA
      asm volatile("tensor_load_to_lds %0, %1" :: "s"(g0), "s"(g1) : "memory");
      asm volatile("s_wait_tensorcnt 0x0" ::: "memory");  // drain before barrier
    }
  }

  // ---- V: staged transposed through VGPRs (DMA cannot transpose) ----
  for (int i = threadIdx.x; i < 64 * HD; i += 128) {
    const int row = i >> 5, col = i & 31;
    const _Float16 vv = (row < N_TOK) ? vh[gbase + (size_t)row * C_DIM + col]
                                      : (_Float16)0.f;
    Vt[col * 64 + row] = vv;
  }

  asm volatile("s_wait_asynccnt 0x0" ::: "memory");
#if __has_builtin(__builtin_amdgcn_cluster_id_x) && __has_builtin(__builtin_amdgcn_s_cluster_barrier)
  if (__builtin_amdgcn_cluster_id_x() != 0)   // uniform; NOP unless clustered
    __builtin_amdgcn_s_cluster_barrier();
#endif
  __syncthreads();

  const int lane = threadIdx.x & 31;
  const int w    = threadIdx.x >> 5;      // wave -> 16-row strip
  const int hi   = lane >> 4;
  const int l15  = lane & 15;

  // ---- S = Q @ K^T : strip of 16 rows per wave, 4 column tiles ----
  {
    v16h a;
    const _Float16* pa = Qs + (16 * w + l15) * HD + 16 * hi;
#pragma unroll
    for (int j = 0; j < 8; ++j) { a[j] = pa[j]; a[8 + j] = pa[8 + j]; }
#pragma unroll
    for (int ntile = 0; ntile < 4; ++ntile) {
      const v16h b = *(const v16h*)(Ks + (16 * ntile + l15) * HD + 16 * hi);
      v8f s = {};
      s = wmma_f16(a, b, s);
#pragma unroll
      for (int r = 0; r < 8; ++r)
        Ss[(16 * w + r + 8 * hi) * 64 + 16 * ntile + l15] = s[r];
    }
  }
  __syncthreads();

  // ---- softmax(S + rel_bias); Swin index: bias_table[c(i)+c(48-j)][h] ----
  {
    const int row = threadIdx.x;
    if (row < 64) {
      if (row >= N_TOK) {
        for (int m = 0; m < 64; ++m) Ps[row * 64 + m] = (_Float16)0.f;
      } else {
        const int ci = 13 * (row / 7) + row % 7;
        float mx = -1e30f;
        for (int m = 0; m < N_TOK; ++m) {
          const int t  = 48 - m;
          const int cj = 13 * (t / 7) + t % 7;
          const float sv = Ss[row * 64 + m] + bias_table[(ci + cj) * NHEADS + head];
          Ss[row * 64 + m] = sv;
          mx = fmaxf(mx, sv);
        }
        float sum = 0.f;
        for (int m = 0; m < N_TOK; ++m) {
          const float e = __expf(Ss[row * 64 + m] - mx);
          Ss[row * 64 + m] = e;
          sum += e;
        }
        const float inv = 1.f / sum;
        for (int m = 0; m < N_TOK; ++m) Ps[row * 64 + m] = (_Float16)(Ss[row * 64 + m] * inv);
        for (int m = N_TOK; m < 64; ++m) Ps[row * 64 + m] = (_Float16)0.f;
      }
    }
  }
  __syncthreads();

  // ---- O = P @ V : 64x32, 2 k-steps, 2 column tiles per wave ----
  {
    v8f o0 = {}, o1 = {};
    for (int kk0 = 0; kk0 < 64; kk0 += 32) {
      v16h a;
      const _Float16* pa = Ps + (16 * w + l15) * 64 + kk0 + 16 * hi;
#pragma unroll
      for (int j = 0; j < 8; ++j) { a[j] = pa[j]; a[8 + j] = pa[8 + j]; }
      const v16h b0 = *(const v16h*)(Vt + l15 * 64 + kk0 + 16 * hi);
      const v16h b1 = *(const v16h*)(Vt + (16 + l15) * 64 + kk0 + 16 * hi);
      o0 = wmma_f16(a, b0, o0);
      o1 = wmma_f16(a, b1, o1);
    }
#pragma unroll
    for (int r = 0; r < 8; ++r) {
      const int tok = 16 * w + r + 8 * hi;
      if (tok < N_TOK) {
        const size_t orow = ((size_t)bwin * N_TOK + tok) * C_DIM + (size_t)head * HD;
        Xo[orow + l15]      = (_Float16)o0[r];
        Xo[orow + 16 + l15] = (_Float16)o1[r];
      }
    }
  }
}

// ---------------------------------------------------------------------------
// Kernel 3: output projection, f16 X @ f16 Wp^T + bp -> f32. 16x64 per wave.
// ---------------------------------------------------------------------------
__global__ __launch_bounds__(256) void out_proj_wmma(
    const _Float16* __restrict__ X, const _Float16* __restrict__ Whp,
    const float* __restrict__ bp, float* __restrict__ out)
{
  const int lane = threadIdx.x & 31;
  const int wave = threadIdx.x >> 5;
  const int tile = blockIdx.x * WAVES_PER_BLOCK + wave;
  const int mt = tile / NT64, nt = tile % NT64;
  const int m0 = mt * 16, n0 = nt * 64;
  const int hi  = lane >> 4;
  const int l15 = lane & 15;

  const _Float16* arow  = X   + (size_t)(m0 + l15) * C_DIM + 16 * hi;
  const _Float16* brow0 = Whp + (size_t)(n0 +      l15) * C_DIM + 16 * hi;
  const _Float16* brow1 = Whp + (size_t)(n0 + 16 + l15) * C_DIM + 16 * hi;
  const _Float16* brow2 = Whp + (size_t)(n0 + 32 + l15) * C_DIM + 16 * hi;
  const _Float16* brow3 = Whp + (size_t)(n0 + 48 + l15) * C_DIM + 16 * hi;

  v8f acc0 = {}, acc1 = {}, acc2 = {}, acc3 = {};
  for (int k0 = 0; k0 < C_DIM; k0 += 32) {
    __builtin_prefetch(arow + k0 + 32, 0, 3);
    const v8h a0 = *(const v8h*)(arow + k0);
    const v8h a1 = *(const v8h*)(arow + k0 + 8);
    const v16h a = __builtin_shufflevector(a0, a1, 0, 1, 2, 3, 4, 5, 6, 7,
                                                   8, 9, 10, 11, 12, 13, 14, 15);
    acc0 = wmma_f16(a, *(const v16h*)(brow0 + k0), acc0);
    acc1 = wmma_f16(a, *(const v16h*)(brow1 + k0), acc1);
    acc2 = wmma_f16(a, *(const v16h*)(brow2 + k0), acc2);
    acc3 = wmma_f16(a, *(const v16h*)(brow3 + k0), acc3);
  }

#pragma unroll
  for (int t = 0; t < 4; ++t) {
    const v8f accv = (t == 0) ? acc0 : (t == 1) ? acc1 : (t == 2) ? acc2 : acc3;
    const int col = n0 + 16 * t + l15;
    const float bval = bp[col];
#pragma unroll
    for (int r = 0; r < 8; ++r) {
      const int m = m0 + r + 8 * hi;
      out[(size_t)m * C_DIM + col] = accv[r] + bval;
    }
  }
}

// ---------------------------------------------------------------------------
extern "C" void kernel_launch(void* const* d_in, const int* in_sizes, int n_in,
                              void* d_out, int out_size, void* d_ws, size_t ws_size,
                              hipStream_t stream) {
  const float* q  = (const float*)d_in[0];
  const float* k  = (const float*)d_in[1];
  const float* v  = (const float*)d_in[2];
  const float* Wq = (const float*)d_in[3];
  const float* bq = (const float*)d_in[4];
  const float* Wk = (const float*)d_in[5];
  const float* bk = (const float*)d_in[6];
  const float* Wv = (const float*)d_in[7];
  const float* bv = (const float*)d_in[8];
  const float* Wp = (const float*)d_in[9];
  const float* bp = (const float*)d_in[10];
  const float* bias_table = (const float*)d_in[11];
  float* out = (float*)d_out;

  // Workspace: Wh(4 matrices f16) | qh | kh | vh | Xo (each M*C f16)
  const size_t elems = (size_t)M_TOTAL * C_DIM;
  _Float16* Wh = (_Float16*)d_ws;
  _Float16* qh = Wh + (size_t)4 * WELEMS;
  _Float16* kh = qh + elems;
  _Float16* vh = kh + elems;
  _Float16* Xo = vh + elems;

  weights_to_f16<<<dim3(WELEMS / 256, 1, 4), 256, 0, stream>>>(Wq, Wk, Wv, Wp, Wh);
  qkv_proj_wmma<<<dim3(GEMM_BLOCKS, 1, 3), 256, 0, stream>>>(
      q, k, v, Wh, bq, bk, bv, qh, kh, vh);
  win_attn_wmma<<<dim3(B_WIN * NHEADS), 128, 0, stream>>>(
      qh, kh, vh, bias_table, Xo);
  out_proj_wmma<<<dim3(GEMM_BLOCKS), 256, 0, stream>>>(Xo, Wh + (size_t)3 * WELEMS, bp, out);
}
